// ConvVectorQuantizer_24094766531143
// MI455X (gfx1250) — compile-verified, matrix operand order checked by reading, not run on hardware
//
#include <hip/hip_runtime.h>
#include <hip/hip_bf16.h>

// ---------------------------------------------------------------------------
// VQ-VAE vector quantizer for MI455X (gfx1250, wave32, WMMA).
//
//   z_e:      (32,256,32,32) f32      codebook: (1024,256) f32
//   outputs:  e_k (B,C,H,W) f32 | e_k_st (B,C,H,W) f32 | ids (B,H,W) as f32
//
// Distance matmul (32768 x 1024 x 256) via v_wmma_f32_16x16x32_bf16 with a
// hi/lo bf16 split of both operands (3 WMMAs per 32-wide K chunk) for ~fp32
// argmin accuracy on the fast bf16 matrix pipe. A-tiles (codebook, L2
// resident) are double-buffered so their loads overlap the WMMA bursts.
// ---------------------------------------------------------------------------

#define CB_K   1024
#define CB_D   256
#define NB     32
#define HWSZ   1024                 // H*W
#define NVEC   (NB * HWSZ)          // 32768 rows
#define ELEMS  (NB * CB_D * HWSZ)   // 8388608 per output tensor

typedef __attribute__((ext_vector_type(16))) __bf16 v16bf;
typedef __attribute__((ext_vector_type(8)))  float  v8f;

union BF16x16 { v16bf v; unsigned int u[8]; };

static __device__ __forceinline__ unsigned short f2bf(float f) {
  unsigned u = __float_as_uint(f);
  u += 0x7FFFu + ((u >> 16) & 1u);       // round-to-nearest-even
  return (unsigned short)(u >> 16);
}
static __device__ __forceinline__ float bf2f(unsigned short h) {
  return __uint_as_float(((unsigned)h) << 16);
}

static __device__ __forceinline__ void load_atile(const unsigned short* __restrict__ cb_hi,
                                                  const unsigned short* __restrict__ cb_lo,
                                                  size_t off, BF16x16& ah, BF16x16& al) {
  const uint4* ph = (const uint4*)(cb_hi + off);
  const uint4* pl = (const uint4*)(cb_lo + off);
  uint4 t0 = ph[0], t1 = ph[2];          // +2 uint4 = +16 elements (K+16)
  uint4 s0 = pl[0], s1 = pl[2];
  ah.u[0]=t0.x; ah.u[1]=t0.y; ah.u[2]=t0.z; ah.u[3]=t0.w;
  ah.u[4]=t1.x; ah.u[5]=t1.y; ah.u[6]=t1.z; ah.u[7]=t1.w;
  al.u[0]=s0.x; al.u[1]=s0.y; al.u[2]=s0.z; al.u[3]=s0.w;
  al.u[4]=s1.x; al.u[5]=s1.y; al.u[6]=s1.z; al.u[7]=s1.w;
}

// ---------------------------------------------------------------------------
// Phase 0: codebook row norms + hi/lo bf16 split of the codebook.
// ---------------------------------------------------------------------------
__global__ void vq_prep(const float* __restrict__ cb,
                        float* __restrict__ wnorm,
                        unsigned short* __restrict__ cb_hi,
                        unsigned short* __restrict__ cb_lo) {
  const int k = blockIdx.x;          // 0..1023
  const int t = threadIdx.x;         // 0..255
  const float x = cb[k * CB_D + t];
  const unsigned short hb = f2bf(x);
  const unsigned short lb = f2bf(x - bf2f(hb));
  cb_hi[k * CB_D + t] = hb;
  cb_lo[k * CB_D + t] = lb;

  __shared__ float red[256];
  red[t] = x * x;
  __syncthreads();
  #pragma unroll
  for (int s = 128; s > 0; s >>= 1) {
    if (t < s) red[t] += red[t + s];
    __syncthreads();
  }
  if (t == 0) wnorm[k] = red[0];
}

// ---------------------------------------------------------------------------
// Phase 1: per-row argmin over 1024 codes via WMMA.
// One wave owns 16 rows; its bf16-split z panel (16 n x 256 c, hi+lo) lives
// in 128 VGPRs; loops 64 k-tiles streaming the split codebook from L2 with
// double-buffered A tiles. wnorm is staged in LDS (ds_read per tile).
// ---------------------------------------------------------------------------
__global__ __launch_bounds__(256, 1)
void vq_argmin(const float* __restrict__ z,
               const unsigned short* __restrict__ cb_hi,
               const unsigned short* __restrict__ cb_lo,
               const float* __restrict__ wnorm,
               int*   __restrict__ ids,
               float* __restrict__ ids_f) {
  const int lane = threadIdx.x & 31;
  const int wid  = threadIdx.x >> 5;
  const int gw   = blockIdx.x * 8 + wid;   // 0..2047
  const int n0   = gw << 4;                // first of 16 rows
  const int bb   = n0 >> 10;               // batch index (HW = 1024)
  const int hw0  = n0 & (HWSZ - 1);
  const int h    = (lane >> 4) & 1;        // lane half (WMMA layout)
  const int nl   = lane & 15;

  // Stage wnorm (4KB) into LDS once per block.
  __shared__ float s_wnorm[CB_K];
  #pragma unroll
  for (int i = 0; i < 4; ++i) s_wnorm[threadIdx.x + 256 * i] = wnorm[threadIdx.x + 256 * i];
  __syncthreads();

  // z_e[b][c][hw]: row n (fixed b,hw), varying c has stride HW floats.
  const float* zb = z + (size_t)bb * (CB_D * HWSZ) + hw0 + nl;

  // ---- Load + split B panel (z^T tiles) into registers ----
  // 16-bit B operand layout (mirror of A): lane half h, VGPR v holds
  // K-pair c = (v&3)*2 + h*8 + (v>>2)*16 (+ 32*chunk), N = lane%16.
  BF16x16 bh[8], bl[8];
  #pragma unroll
  for (int cc = 0; cc < 8; ++cc) {
    #pragma unroll
    for (int v = 0; v < 8; ++v) {
      const int c0 = cc * 32 + (v & 3) * 2 + h * 8 + ((v >> 2) << 4);
      const float f0 = zb[(size_t)c0 * HWSZ];
      const float f1 = zb[(size_t)(c0 + 1) * HWSZ];
      const unsigned short h0 = f2bf(f0), h1 = f2bf(f1);
      const unsigned short l0 = f2bf(f0 - bf2f(h0));
      const unsigned short l1 = f2bf(f1 - bf2f(h1));
      bh[cc].u[v] = (unsigned)h0 | ((unsigned)h1 << 16);
      bl[cc].u[v] = (unsigned)l0 | ((unsigned)l1 << 16);
    }
  }

  float best = 3.0e38f;
  int   bidx = 0;
  const int kr = nl;                       // A-matrix row within k-tile

  // A-tile element offset for flat iteration it = kt*8+cc:
  //   (kt*16 + kr)*256 + cc*32 + h*8  =  base + kt*4096 + cc*32
  const size_t abase = (size_t)kr * CB_D + h * 8;

  BF16x16 ah[2], al[2];
  load_atile(cb_hi, cb_lo, abase, ah[0], al[0]);   // prologue: (kt=0, cc=0)

  for (int kt = 0; kt < 64; ++kt) {
    v8f acc = {};
    #pragma unroll
    for (int cc = 0; cc < 8; ++cc) {
      const int cur = cc & 1, nxt = cur ^ 1;
      // Prefetch next chunk (wraps harmlessly at the very end).
      const int nit = (kt * 8 + cc + 1) & 511;
      load_atile(cb_hi, cb_lo,
                 abase + (size_t)(nit >> 3) * 4096 + (size_t)(nit & 7) * 32,
                 ah[nxt], al[nxt]);
      // dot = hi*hi + hi*lo + lo*hi (lo*lo negligible), f32 accumulate
      acc = __builtin_amdgcn_wmma_f32_16x16x32_bf16(
                false, ah[cur].v, false, bh[cc].v, (short)0, acc, false, false);
      acc = __builtin_amdgcn_wmma_f32_16x16x32_bf16(
                false, ah[cur].v, false, bl[cc].v, (short)0, acc, false, false);
      acc = __builtin_amdgcn_wmma_f32_16x16x32_bf16(
                false, al[cur].v, false, bh[cc].v, (short)0, acc, false, false);
    }

    // D layout: VGPR v, lane half h  <->  k = k0 + 8*h + v.
    const int k0 = kt << 4;
    const float4* wp = (const float4*)(s_wnorm + k0 + h * 8);
    const float4 w0 = wp[0], w1 = wp[1];
    const float wv[8] = {w0.x, w0.y, w0.z, w0.w, w1.x, w1.y, w1.z, w1.w};
    #pragma unroll
    for (int v = 0; v < 8; ++v) {
      const float s = wv[v] - 2.0f * acc[v]; // ||z||^2 row-constant: dropped
      const int  kk = k0 + h * 8 + v;
      if (s < best) { best = s; bidx = kk; } // strict < keeps lowest k on tie
    }
  }

  // Combine lane L with L+16 (SWAPX16 via ds_swizzle, wave32).
  const int ovi = __builtin_amdgcn_ds_swizzle(__float_as_int(best), 0x401F);
  const int oii = __builtin_amdgcn_ds_swizzle(bidx, 0x401F);
  const float ov = __int_as_float(ovi);
  if (ov < best || (ov == best && oii < bidx)) { best = ov; bidx = oii; }

  if (h == 0) {
    ids[n0 + nl]   = bidx;
    ids_f[n0 + nl] = (float)bidx;
  }
}

// ---------------------------------------------------------------------------
// Phase 2: gather codebook rows + straight-through output (bandwidth bound).
// ---------------------------------------------------------------------------
__global__ void vq_gather(const float* __restrict__ z,
                          const float* __restrict__ cb,
                          const int*   __restrict__ ids,
                          float* __restrict__ ek,
                          float* __restrict__ ekst) {
  const size_t idx = (size_t)blockIdx.x * blockDim.x + threadIdx.x;
  const int hw = (int)(idx & (HWSZ - 1));
  const int c  = (int)((idx >> 10) & (CB_D - 1));
  const int bb = (int)(idx >> 18);              // C*HW = 2^18
  const int n  = (bb << 10) | hw;
  const int id = ids[n];
  const float e  = cb[(size_t)id * CB_D + c];   // 1MB codebook: L2 resident
  const float zz = z[idx];
  ek[idx]   = e;
  ekst[idx] = zz + (e - zz);                    // exact STE forward expression
}

// ---------------------------------------------------------------------------
extern "C" void kernel_launch(void* const* d_in, const int* in_sizes, int n_in,
                              void* d_out, int out_size, void* d_ws, size_t ws_size,
                              hipStream_t stream) {
  (void)in_sizes; (void)n_in; (void)out_size; (void)ws_size;
  const float* z  = (const float*)d_in[0];
  const float* cb = (const float*)d_in[1];

  float* out  = (float*)d_out;
  float* ek   = out;                    // (B,C,H,W)
  float* ekst = out + (size_t)ELEMS;    // (B,C,H,W)
  float* idsf = out + (size_t)2 * ELEMS;// (B,H,W) as float

  char* ws = (char*)d_ws;               // ~1.13 MB used
  float*          wnorm = (float*)ws;                                 //   4 KB
  unsigned short* cb_hi = (unsigned short*)(ws + 4096);               // 512 KB
  unsigned short* cb_lo = (unsigned short*)(ws + 4096 + 524288);      // 512 KB
  int*            ids   = (int*)(ws + 4096 + 2 * 524288);             // 128 KB

  vq_prep  <<<CB_K, 256, 0, stream>>>(cb, wnorm, cb_hi, cb_lo);
  vq_argmin<<<NVEC / 16 / 8, 256, 0, stream>>>(z, cb_hi, cb_lo, wnorm, ids, idsf);
  vq_gather<<<ELEMS / 256, 256, 0, stream>>>(z, cb, ids, ek, ekst);
}
